// DoraLinearLayer_74577812128283
// MI455X (gfx1250) — matrix operand order repaired and checked by reading, not compiled, not run on hardware
//
#include <hip/hip_runtime.h>
#include <cstdint>
#include <cstddef>

// ---------------------------------------------------------------------------
// Problem constants (from reference): B=8, S=2048, D_IN=D_OUT=4096, R=16
// ---------------------------------------------------------------------------
static constexpr int kM    = 8 * 2048;   // rows of x (B*S) = 16384
static constexpr int kK    = 4096;       // D_IN
static constexpr int kN    = 4096;       // D_OUT
static constexpr int kR    = 16;         // rank

typedef __bf16 bf16_t;
typedef __attribute__((ext_vector_type(8)))  bf16_t v8bf;
typedef __attribute__((ext_vector_type(16))) bf16_t v16bf;
typedef __attribute__((ext_vector_type(8)))  float  v8f;
typedef __attribute__((ext_vector_type(4)))  float  v4f;

union V16Pair { v16bf v; v8bf h[2]; };

// ---------------------------------------------------------------------------
// Kernel 1: fp32 x  ->  bf16 xb   (exact coverage, 8 elems / thread)
// x is read exactly once -> non-temporal loads (TH=NT keeps L2 clean for xb).
// xb writes stay regular-temporal so they allocate in L2 for the GEMM.
// ---------------------------------------------------------------------------
__global__ __launch_bounds__(256) void k_cvt_bf16(const float* __restrict__ x,
                                                  bf16_t* __restrict__ xb) {
  size_t i = ((size_t)blockIdx.x * 256 + threadIdx.x) * 8;
  v4f a = __builtin_nontemporal_load((const v4f*)(x + i));
  v4f b = __builtin_nontemporal_load((const v4f*)(x + i + 4));
  v8bf o;
  o[0] = (bf16_t)a[0]; o[1] = (bf16_t)a[1]; o[2] = (bf16_t)a[2]; o[3] = (bf16_t)a[3];
  o[4] = (bf16_t)b[0]; o[5] = (bf16_t)b[1]; o[6] = (bf16_t)b[2]; o[7] = (bf16_t)b[3];
  *(v8bf*)(xb + i) = o;
}

// ---------------------------------------------------------------------------
// Kernel 2: one block per output row o.
//   c[r]      = U[o,r]*S[r]*O[o]
//   adapter_i = sum_r c[r]*V[r,i]
//   scale     = magnitude[o] / || base[o,:] + adapter ||
//   W_eff[o,i]= (scale-1)*base[o,i] + scale*adapter_i   (stored bf16)
// ---------------------------------------------------------------------------
__global__ __launch_bounds__(256) void k_build_weff(
    const float* __restrict__ U, const float* __restrict__ S,
    const float* __restrict__ O, const float* __restrict__ mag,
    const float* __restrict__ base, const float* __restrict__ V,
    bf16_t* __restrict__ weff) {
  const int o   = blockIdx.x;
  const int tid = threadIdx.x;

  __shared__ float c[kR];
  __shared__ float red[256];
  __shared__ float scale_sh;

  if (tid < kR) c[tid] = U[(size_t)o * kR + tid] * S[tid] * O[o];
  __syncthreads();

  const float* brow = base + (size_t)o * kK;
  float bv[kK / 256];
  float av[kK / 256];
  float ss = 0.0f;
#pragma unroll
  for (int j = 0; j < kK / 256; ++j) {
    const int i = tid + j * 256;
    float ad = 0.0f;
#pragma unroll
    for (int r = 0; r < kR; ++r) ad += c[r] * V[(size_t)r * kK + i];
    const float b = __builtin_nontemporal_load(brow + i);  // base read once
    bv[j] = b;
    av[j] = ad;
    const float w = b + ad;
    ss += w * w;
  }

  red[tid] = ss;
  __syncthreads();
  for (int s = 128; s > 0; s >>= 1) {
    if (tid < s) red[tid] += red[tid + s];
    __syncthreads();
  }
  if (tid == 0) scale_sh = mag[o] / sqrtf(red[0]);
  __syncthreads();

  const float sc  = scale_sh;
  const float scm = sc - 1.0f;
  bf16_t* wrow = weff + (size_t)o * kK;
#pragma unroll
  for (int j = 0; j < kK / 256; ++j) {
    const int i = tid + j * 256;
    wrow[i] = (bf16_t)(scm * bv[j] + sc * av[j]);  // RT: keep weff in L2
  }
}

// ---------------------------------------------------------------------------
// Kernel 3: out[M,N] = xb[M,K] @ weff[N,K]^T  via v_wmma_f32_16x16x32_bf16.
// Workgroup: 8 waves, 64(M) x 256(N) tile; wave: 32x64 = 2x4 WMMA tiles.
// Output stores are non-temporal: out (256 MiB) streams to HBM once and must
// not evict the L2-resident operands (xb 128 MiB + weff 32 MiB < 192 MB L2).
// ---------------------------------------------------------------------------
__global__ __launch_bounds__(256) void k_gemm(const bf16_t* __restrict__ xb,
                                              const bf16_t* __restrict__ weff,
                                              float* __restrict__ out) {
  const int wave  = threadIdx.x >> 5;   // 0..7
  const int lane  = threadIdx.x & 31;
  const int half  = lane >> 4;          // 0/1: K-interleave group (ISA layout)
  const int row   = lane & 15;

  const int waveM = wave & 1;           // 0..1
  const int waveN = wave >> 1;          // 0..3
  const int m0 = blockIdx.x * 64  + waveM * 32;
  const int n0 = blockIdx.y * 256 + waveN * 64;

  // A-fragment base pointers (ISA 16-bit A 16x32 layout):
  // lane<16  holds row `row`, K chunks {k..k+7} and {k+16..k+23}
  // lane>=16 holds row `row`, K chunks {k+8..k+15} and {k+24..k+31}
  const bf16_t* aBase[2];
  aBase[0] = xb + (size_t)(m0 + row)      * kK + (half ? 8 : 0);
  aBase[1] = xb + (size_t)(m0 + 16 + row) * kK + (half ? 8 : 0);

  // B-fragment base pointers: lane group holds column `row`,
  // K chunks {k..k+7},{k+16..k+23} (half=0) or {k+8..k+15},{k+24..k+31} (half=1).
  const bf16_t* bBase[4];
#pragma unroll
  for (int j = 0; j < 4; ++j)
    bBase[j] = weff + (size_t)(n0 + j * 16 + row) * kK + (half ? 8 : 0);

  v8f acc[2][4] = {};

#pragma unroll 2
  for (int k = 0; k < kK; k += 32) {
    v16bf a[2], b[4];
#pragma unroll
    for (int t = 0; t < 2; ++t) {
      V16Pair p;
      p.h[0] = *(const v8bf*)(aBase[t] + k);
      p.h[1] = *(const v8bf*)(aBase[t] + k + 16);
      a[t] = p.v;
    }
#pragma unroll
    for (int j = 0; j < 4; ++j) {
      V16Pair q;
      q.h[0] = *(const v8bf*)(bBase[j] + k);
      q.h[1] = *(const v8bf*)(bBase[j] + k + 16);
      b[j] = q.v;
    }

#pragma unroll
    for (int t = 0; t < 2; ++t)
#pragma unroll
      for (int j = 0; j < 4; ++j)
        acc[t][j] = __builtin_amdgcn_wmma_f32_16x16x32_bf16(
            /*neg_a=*/false, a[t], /*neg_b=*/false, b[j],
            /*c_mod=*/(short)0, acc[t][j],
            /*reuse_a=*/false, /*reuse_b=*/false);
  }

  // D layout: lanes 0-15 -> N=lane, M=vgpr; lanes 16-31 -> N=lane-16, M=vgpr+8
#pragma unroll
  for (int t = 0; t < 2; ++t) {
#pragma unroll
    for (int j = 0; j < 4; ++j) {
      float* o = out + (size_t)(m0 + t * 16 + half * 8) * kN + (n0 + j * 16 + row);
#pragma unroll
      for (int v = 0; v < 8; ++v)
        __builtin_nontemporal_store(acc[t][j][v], o + (size_t)v * kN);
    }
  }
}

// ---------------------------------------------------------------------------
// Launch
// ---------------------------------------------------------------------------
extern "C" void kernel_launch(void* const* d_in, const int* in_sizes, int n_in,
                              void* d_out, int out_size, void* d_ws, size_t ws_size,
                              hipStream_t stream) {
  const float* x    = (const float*)d_in[0];
  const float* U    = (const float*)d_in[1];
  const float* V    = (const float*)d_in[2];
  const float* S    = (const float*)d_in[3];
  const float* O    = (const float*)d_in[4];
  const float* mag  = (const float*)d_in[5];
  const float* base = (const float*)d_in[6];
  float* out = (float*)d_out;

  // Workspace layout: xb (M*K bf16 = 128 MiB) | weff (N*K bf16 = 32 MiB).
  // Both GEMM operands fit in the 192 MB L2 together.
  bf16_t* xb   = (bf16_t*)d_ws;
  bf16_t* weff = (bf16_t*)((char*)d_ws + (size_t)kM * kK * sizeof(bf16_t));

  // 1) x -> bf16
  k_cvt_bf16<<<dim3(((size_t)kM * kK) / (8 * 256)), 256, 0, stream>>>(x, xb);
  // 2) fold rank-16 adapter + DoRA norm/scale into effective bf16 weights
  k_build_weff<<<dim3(kN), 256, 0, stream>>>(U, S, O, mag, base, V, weff);
  // 3) single big bf16 WMMA GEMM
  k_gemm<<<dim3(kM / 64, kN / 256), 256, 0, stream>>>(xb, weff, out);
}
